// CTLSTM_90847148245598
// MI455X (gfx1250) — matrix operand order repaired and checked by reading, not compiled, not run on hardware
//
#include <hip/hip_runtime.h>
#include <cmath>

typedef __attribute__((ext_vector_type(4)))  float        v4f;
typedef __attribute__((ext_vector_type(8)))  float        v8f;
typedef __attribute__((ext_vector_type(2)))  unsigned int v2u;
typedef __attribute__((ext_vector_type(4)))  unsigned int v4u;
typedef __attribute__((ext_vector_type(8)))  __bf16       v8bf;
typedef __attribute__((ext_vector_type(16))) __bf16       v16bf;

#define BATCH   1048576
#define NTILES  (BATCH / 16)   // 65536
#define BLOCK   128            // 4 waves (wave32)
#define WAVES   4
#define GRID    4096           // 4096*4 wave-slots -> exactly 4 tiles/wave (uniform barriers)
#define KP      104            // padded K row stride in bf16 elems (208 B, 16B-aligned, conflict-free)

__device__ __forceinline__ float fast_tanhf(float x) {
#if __has_builtin(__builtin_amdgcn_tanhf)
    return __builtin_amdgcn_tanhf(x);           // v_tanh_f32
#elif __has_builtin(__builtin_amdgcn_tanh_f32)
    return __builtin_amdgcn_tanh_f32(x);
#else
    return tanhf(x);
#endif
}

// Split two fp32 values into packed bf16 words: result[0] = {hi1,hi0}, result[1] = {lo1,lo0}.
// hi = truncation (exact top 16 bits), lo = bf16 of the exact residual.
__device__ __forceinline__ v2u split2(float a0, float a1) {
    unsigned int b0 = __float_as_uint(a0), b1 = __float_as_uint(a1);
    unsigned int h0 = b0 & 0xffff0000u,    h1 = b1 & 0xffff0000u;
    float r0 = a0 - __uint_as_float(h0);
    float r1 = a1 - __uint_as_float(h1);
    v2u r;
    r[0] = (h0 >> 16) | h1;
    r[1] = (__float_as_uint(r0) >> 16) | (__float_as_uint(r1) & 0xffff0000u);
    return r;
}

__device__ __forceinline__ v16bf cat16(v8bf a, v8bf b) {
    return __builtin_shufflevector(a, b, 0,1,2,3,4,5,6,7,8,9,10,11,12,13,14,15);
}

__global__ __launch_bounds__(BLOCK)
void ctlstm_wmma_bf16x3_kernel(
    const float* __restrict__ x,      const float* __restrict__ h_prev,
    const float* __restrict__ c_prev, const float* __restrict__ dt,
    const float* __restrict__ Wi, const float* __restrict__ bi,
    const float* __restrict__ Wf, const float* __restrict__ bf,
    const float* __restrict__ Wo, const float* __restrict__ bo,
    const float* __restrict__ Wc, const float* __restrict__ bc,
    const float* __restrict__ Wd, const float* __restrict__ bd,
    float* __restrict__ out)
{
    // W transposed: wT[prec][c*KP + k], c = g*16+h (output col), k = feature. 33.3 KB
    __shared__ unsigned short wT[2][80 * KP];
    // per-wave combined tile [m][k] hi/lo. 26.6 KB
    __shared__ unsigned short atl[WAVES][2][16 * KP];

    const int tid  = threadIdx.x;
    const int wave = tid >> 5;
    const int lane = tid & 31;
    const int half = lane >> 4;
    const int l15  = lane & 15;

    // ---- stage W (5 gates of [80 x 16]) as transposed bf16 hi/lo ----
    {
        const float* Wg[5] = { Wi, Wf, Wo, Wc, Wd };
        #pragma unroll
        for (int g = 0; g < 5; ++g) {
            const float* W = Wg[g];
            for (int idx = tid; idx < 80 * 16; idx += BLOCK) {
                int k = idx >> 4, h = idx & 15, c = g * 16 + h;
                float a = W[idx];
                unsigned int b = __float_as_uint(a);
                unsigned int hb = b & 0xffff0000u;
                float res = a - __uint_as_float(hb);
                wT[0][c * KP + k] = (unsigned short)(hb >> 16);
                wT[1][c * KP + k] = (unsigned short)(__float_as_uint(res) >> 16);
            }
        }
        // zero-pad K = 80..95 (read by the t=2 WMMA step): 80 cols x 2 prec x 2 chunks
        for (int u = tid; u < 80 * 4; u += BLOCK) {
            int c = u >> 2, prec = (u >> 1) & 1, ch = u & 1;
            *(v4u*)&wT[prec][c * KP + 80 + ch * 8] = (v4u)0u;
        }
    }
    // zero-pad per-wave A tiles K = 80..95 (done once; never overwritten)
    for (int u = lane; u < 16 * 4; u += 32) {
        int m = u >> 2, prec = (u >> 1) & 1, ch = u & 1;
        *(v4u*)&atl[wave][prec][m * KP + 80 + ch * 8] = (v4u)0u;
    }
    // per-lane bias (C/D layout: N = lane&15, same across the 8 acc VGPRs)
    const float biasv[5] = { bi[l15], bf[l15], bo[l15], bc[l15], bd[l15] };
    __syncthreads();

    unsigned short* ath = atl[wave][0];
    unsigned short* alo = atl[wave][1];

    for (int tile = blockIdx.x * WAVES + wave; tile < NTILES; tile += GRID * WAVES) {
        // ---- stage combined = [x | h] (16 rows) as bf16 hi/lo into LDS ----
        {
            const float* xb = x + (size_t)tile * (16 * 64);      // 4 KB contiguous
            #pragma unroll
            for (int it = 0; it < 8; ++it) {
                int flat = (lane + it * 32) * 4;                 // 0..1023
                int row = flat >> 6, col = flat & 63;
                v4f v = *(const v4f*)(xb + flat);
                v2u p01 = split2(v[0], v[1]);
                v2u p23 = split2(v[2], v[3]);
                v2u hw; hw[0] = p01[0]; hw[1] = p23[0];
                v2u lw; lw[0] = p01[1]; lw[1] = p23[1];
                *(v2u*)&ath[row * KP + col] = hw;                // 8B aligned (col%4==0)
                *(v2u*)&alo[row * KP + col] = lw;
            }
            const float* hb = h_prev + (size_t)tile * (16 * 16); // 1 KB contiguous
            #pragma unroll
            for (int it = 0; it < 2; ++it) {
                int flat = (lane + it * 32) * 4;                 // 0..255
                int row = flat >> 4, col = flat & 15;
                v4f v = *(const v4f*)(hb + flat);
                v2u p01 = split2(v[0], v[1]);
                v2u p23 = split2(v[2], v[3]);
                v2u hw; hw[0] = p01[0]; hw[1] = p23[0];
                v2u lw; lw[0] = p01[1]; lw[1] = p23[1];
                *(v2u*)&ath[row * KP + 64 + col] = hw;
                *(v2u*)&alo[row * KP + 64 + col] = lw;
            }
        }
        __syncthreads();

        // ---- z[16 x 80] = combined @ W + b, bf16x3: hi*hi + lo*hi + hi*lo ----
        v8f acc[5];
        #pragma unroll
        for (int g = 0; g < 5; ++g) {
            v8f a;
            #pragma unroll
            for (int r = 0; r < 8; ++r) a[r] = biasv[g];
            acc[g] = a;
        }
        #pragma unroll
        for (int t = 0; t < 3; ++t) {                 // K tiles of 32 (K padded 80->96)
            // A 16x32 bf16 fragment: lane half 0: K 0-7 & 16-23; half 1: K 8-15 & 24-31
            const int ab = l15 * KP + t * 32 + half * 8;
            v16bf Ahi = cat16(*(const v8bf*)&ath[ab], *(const v8bf*)&ath[ab + 16]);
            v16bf Alo = cat16(*(const v8bf*)&alo[ab], *(const v8bf*)&alo[ab + 16]);
            #pragma unroll
            for (int g = 0; g < 5; ++g) {             // N tiles (= gates)
                // B 32x16 fragment: lane half 0: K 0-7 & 16-23; half 1: K 8-15 & 24-31 of its output column
                const int bb = (g * 16 + l15) * KP + t * 32 + half * 8;
                v16bf Bhi = cat16(*(const v8bf*)&wT[0][bb], *(const v8bf*)&wT[0][bb + 16]);
                v16bf Blo = cat16(*(const v8bf*)&wT[1][bb], *(const v8bf*)&wT[1][bb + 16]);
                acc[g] = __builtin_amdgcn_wmma_f32_16x16x32_bf16(
                    false, Ahi, false, Bhi, (short)0, acc[g], false, false);
                acc[g] = __builtin_amdgcn_wmma_f32_16x16x32_bf16(
                    false, Alo, false, Bhi, (short)0, acc[g], false, false);
                acc[g] = __builtin_amdgcn_wmma_f32_16x16x32_bf16(
                    false, Ahi, false, Blo, (short)0, acc[g], false, false);
            }
        }

        // ---- elementwise epilogue; C/D layout: M = r + 8*half, N = lane&15 ----
        #pragma unroll
        for (int r = 0; r < 8; ++r) {
            int row = tile * 16 + r + 8 * half;
            float zi = acc[0][r], zf = acc[1][r], zo = acc[2][r];
            float zc = acc[3][r], zd = acc[4][r];
            float gi = fast_tanhf(zi);
            float gf = fast_tanhf(zf);
            float go = fast_tanhf(zo);
            float ct = fast_tanhf(zc);
            // stable softplus via hardware exp/log
            float sp = fmaxf(zd, 0.0f) + __logf(1.0f + __expf(-fabsf(zd)));
            float cdec = c_prev[(size_t)row * 16 + l15] * __expf(-sp * dt[row]);
            float cn = gf * cdec + gi * ct;
            float hn = go * fast_tanhf(cn);
            out[(size_t)row * 16 + l15] = hn;                          // h_next
            out[(size_t)BATCH * 16 + (size_t)row * 16 + l15] = cn;     // c_next
        }
        __syncthreads();   // A tiles reused next iteration
    }
}

extern "C" void kernel_launch(void* const* d_in, const int* in_sizes, int n_in,
                              void* d_out, int out_size, void* d_ws, size_t ws_size,
                              hipStream_t stream) {
    (void)in_sizes; (void)n_in; (void)d_ws; (void)ws_size; (void)out_size;
    const float* x      = (const float*)d_in[0];
    const float* h_prev = (const float*)d_in[1];
    const float* c_prev = (const float*)d_in[2];
    const float* dt     = (const float*)d_in[3];
    const float* Wi = (const float*)d_in[4];  const float* bi = (const float*)d_in[5];
    const float* Wf = (const float*)d_in[6];  const float* bf = (const float*)d_in[7];
    const float* Wo = (const float*)d_in[8];  const float* bo = (const float*)d_in[9];
    const float* Wc = (const float*)d_in[10]; const float* bc = (const float*)d_in[11];
    const float* Wd = (const float*)d_in[12]; const float* bd = (const float*)d_in[13];
    float* out = (float*)d_out;

    ctlstm_wmma_bf16x3_kernel<<<GRID, BLOCK, 0, stream>>>(
        x, h_prev, c_prev, dt, Wi, bi, Wf, bf, Wo, bo, Wc, bc, Wd, bd, out);
}
